// TwoLayerRNN_360777253022
// MI455X (gfx1250) — matrix operand order repaired and checked by reading, not compiled, not run on hardware
//
#include <hip/hip_runtime.h>
#include <math.h>

#define MD 512        // M1 == M2
#define TSTEPS 100
#define BATCH 4096
#define BM 16         // batch rows per workgroup
#define LDSTRIDE 520  // bf16 elements per LDS row (512 + 8 pad, keeps 16B row alignment)
#define NWAVES 4
#define NT_PER_WAVE 8 // 512 cols / 16 per tile / 4 waves

typedef __attribute__((ext_vector_type(8)))  float  v8f;
typedef __attribute__((ext_vector_type(8)))  __bf16 v8bf;
typedef __attribute__((ext_vector_type(16))) __bf16 v16bf;

union V16 { v16bf v; v8bf h[2]; };

// Branch-free tanh: native v_tanh if the toolchain exposes it, else
// v_exp_f32 + v_rcp_f32 path (no EXEC-splitting slow path like libm tanhf).
__device__ __forceinline__ float tanh_fast(float x) {
#if __has_builtin(__builtin_amdgcn_tanhf)
  return __builtin_amdgcn_tanhf(x);
#elif __has_builtin(__builtin_amdgcn_tanh_f32)
  return __builtin_amdgcn_tanh_f32(x);
#else
  float ax = __builtin_fabsf(x);
  float e  = __expf(2.0f * ax);              // native v_exp_f32 path
  float r  = 1.0f - 2.0f / (e + 1.0f);       // e==inf -> r==1 (saturated)
  return __builtin_copysignf(r, x);
#endif
}

// Convert the three 512x512 f32 weights to bf16, transposed: out[w][n*512 + k] = W[k][n].
// Makes the WMMA B-operand a contiguous 32-byte read per lane.
__global__ void rnn_prep_kernel(const float* __restrict__ U1,
                                const float* __restrict__ W2,
                                const float* __restrict__ U2,
                                __bf16* __restrict__ out) {
  int idx = blockIdx.x * blockDim.x + threadIdx.x;     // 0 .. 3*512*512-1
  int w = idx >> 18;                                   // weight id
  int r = idx & ((1 << 18) - 1);
  int n = r >> 9;                                      // output row  = source column
  int k = r & (MD - 1);                                // output col  = source row
  const float* src = (w == 0) ? U1 : (w == 1) ? W2 : U2;
  out[((size_t)w << 18) + (size_t)n * MD + k] = (__bf16)src[k * MD + n];
}

__global__ __launch_bounds__(NWAVES * 32)
void rnn_main_kernel(const float* __restrict__ x_seq,   // [B, T]
                     const float* __restrict__ w_in1,   // [512]
                     const float* __restrict__ b1,      // [512]
                     const float* __restrict__ b2,      // [512]
                     const float* __restrict__ w_y,     // [512]
                     const float* __restrict__ b_y,     // [1]
                     const __bf16* __restrict__ u1t,    // [512][512] transposed bf16
                     const __bf16* __restrict__ w2t,
                     const __bf16* __restrict__ u2t,
                     float* __restrict__ out)           // [B]
{
  __shared__ __align__(16) __bf16 h1s[2][BM * LDSTRIDE];
  __shared__ __align__(16) __bf16 h2s[2][BM * LDSTRIDE];
  __shared__ float xs[BM];
  __shared__ float yred[BM];

  const int tid  = threadIdx.x;
  const int lane = tid & 31;
  const int wave = tid >> 5;
  const int b0   = blockIdx.x * BM;

  // Zero initial hidden-state buffers (parity 0); stage x_0; init y accumulators.
  for (int i = tid; i < BM * LDSTRIDE; i += blockDim.x) {
    h1s[0][i] = (__bf16)0.f;
    h2s[0][i] = (__bf16)0.f;
  }
  if (tid < BM) {
    xs[tid]   = x_seq[(b0 + tid) * TSTEPS];
    yred[tid] = b_y[0];
  }
  __syncthreads();

  const int nlo  = lane & 15;   // C: column-in-tile | A: row m | B: column-in-tile
  const int lhi  = lane >> 4;   // C: row-half      | A: K-half| B: K-offset/16
  const int koff = lhi * 16;

  int p = 0;
  for (int t = 0; t < TSTEPS; ++t) {
    // x values for this lane's C rows (m = v + 8*lhi), shared across all n-tiles
    float xv[8];
    #pragma unroll
    for (int v = 0; v < 8; ++v) xv[v] = xs[v + 8 * lhi];

    // ---------------- layer 1: h1 = tanh(x*W_in1 + h1_prev @ U1 + b1) ----------------
    #pragma unroll 1
    for (int nt = 0; nt < NT_PER_WAVE; ++nt) {
      const int n = (wave * NT_PER_WAVE + nt) * 16 + nlo;
      const float w1n = w_in1[n];
      const float b1n = b1[n];
      v8f c;
      #pragma unroll
      for (int v = 0; v < 8; ++v) c[v] = xv[v] * w1n + b1n;   // rank-1 input term

      const __bf16* brow = u1t + (size_t)n * MD + koff;
      const __bf16* arow = &h1s[p][nlo * LDSTRIDE + koff];
      #pragma unroll 4
      for (int k0 = 0; k0 < MD; k0 += 32) {
        V16 a, b;
        a.h[0] = *(const v8bf*)(arow + k0);        // K = k0 + 16*lhi + 0..7
        a.h[1] = *(const v8bf*)(arow + k0 + 8);    // K = k0 + 16*lhi + 8..15
        b.v    = *(const v16bf*)(brow + k0);       // contiguous 32B: W^T[n][k0+koff..+15]
        c = __builtin_amdgcn_wmma_f32_16x16x32_bf16(false, a.v, false, b.v,
                                                    (short)0, c, false, false);
      }
      #pragma unroll
      for (int v = 0; v < 8; ++v) {
        float hv = tanh_fast(c[v]);
        h1s[p ^ 1][(v + 8 * lhi) * LDSTRIDE + n] = (__bf16)hv;
      }
    }
    __syncthreads();

    // ---------------- layer 2: h2 = tanh(h1 @ W_in2 + h2_prev @ U2 + b2) ----------------
    #pragma unroll 1
    for (int nt = 0; nt < NT_PER_WAVE; ++nt) {
      const int n = (wave * NT_PER_WAVE + nt) * 16 + nlo;
      const float b2n = b2[n];
      v8f c;
      #pragma unroll
      for (int v = 0; v < 8; ++v) c[v] = b2n;

      const __bf16* brow2 = w2t + (size_t)n * MD + koff;
      const __bf16* brow3 = u2t + (size_t)n * MD + koff;
      const __bf16* arow1 = &h1s[p ^ 1][nlo * LDSTRIDE + koff];
      const __bf16* arow2 = &h2s[p][nlo * LDSTRIDE + koff];
      #pragma unroll 4
      for (int k0 = 0; k0 < MD; k0 += 32) {
        V16 a, b;
        a.h[0] = *(const v8bf*)(arow1 + k0);
        a.h[1] = *(const v8bf*)(arow1 + k0 + 8);
        b.v    = *(const v16bf*)(brow2 + k0);
        c = __builtin_amdgcn_wmma_f32_16x16x32_bf16(false, a.v, false, b.v,
                                                    (short)0, c, false, false);
      }
      #pragma unroll 4
      for (int k0 = 0; k0 < MD; k0 += 32) {
        V16 a, b;
        a.h[0] = *(const v8bf*)(arow2 + k0);
        a.h[1] = *(const v8bf*)(arow2 + k0 + 8);
        b.v    = *(const v16bf*)(brow3 + k0);
        c = __builtin_amdgcn_wmma_f32_16x16x32_bf16(false, a.v, false, b.v,
                                                    (short)0, c, false, false);
      }
      #pragma unroll
      for (int v = 0; v < 8; ++v) {
        float hv = tanh_fast(c[v]);
        h2s[p ^ 1][(v + 8 * lhi) * LDSTRIDE + n] = (__bf16)hv;
        if (t == TSTEPS - 1) {
          // final readout: y[m] += h2[m][n] * W_y[n], reduced via LDS float atomics
          atomicAdd(&yred[v + 8 * lhi], hv * w_y[n]);
        }
      }
    }

    if (tid < BM && t + 1 < TSTEPS)
      xs[tid] = x_seq[(b0 + tid) * TSTEPS + (t + 1)];
    __syncthreads();
    p ^= 1;
  }

  if (tid < BM) out[b0 + tid] = yred[tid];
}

extern "C" void kernel_launch(void* const* d_in, const int* in_sizes, int n_in,
                              void* d_out, int out_size, void* d_ws, size_t ws_size,
                              hipStream_t stream) {
  const float* x_seq = (const float*)d_in[0];  // (B, T, 1)
  const float* W_in1 = (const float*)d_in[1];  // (1, 512)
  const float* U1    = (const float*)d_in[2];  // (512, 512)
  const float* b1    = (const float*)d_in[3];  // (512,)
  const float* W_in2 = (const float*)d_in[4];  // (512, 512)
  const float* U2    = (const float*)d_in[5];  // (512, 512)
  const float* b2    = (const float*)d_in[6];  // (512,)
  const float* W_y   = (const float*)d_in[7];  // (512, 1)
  const float* b_y   = (const float*)d_in[8];  // (1,)

  __bf16* wsb = (__bf16*)d_ws;                 // 3 * 512*512 bf16 = 1.5 MB
  const __bf16* u1t = wsb;
  const __bf16* w2t = wsb + (size_t)MD * MD;
  const __bf16* u2t = wsb + (size_t)2 * MD * MD;

  rnn_prep_kernel<<<(3 * MD * MD) / 256, 256, 0, stream>>>(U1, W_in2, U2, wsb);
  rnn_main_kernel<<<BATCH / BM, NWAVES * 32, 0, stream>>>(
      x_seq, W_in1, b1, b2, W_y, b_y, u1t, w2t, u2t, (float*)d_out);
}
